// LSHSelfAttention_9062380995185
// MI455X (gfx1250) — 8-bit matrix operand order repaired and checked by reading, compiled, not run on hardware
//
#include <hip/hip_runtime.h>
#include <hip/hip_bf16.h>

typedef float v8f __attribute__((ext_vector_type(8)));
typedef float v2f __attribute__((ext_vector_type(2)));
typedef int   v8i __attribute__((ext_vector_type(8)));

#define TT   2048
#define HH   16
#define DD   64
#define HD   1024
#define NCOL 128

__global__ __launch_bounds__(256) void lsh_rotate_bucket_kernel(
    const float* __restrict__ hidden,
    const float* __restrict__ rot,
    unsigned int* __restrict__ packed_buckets)
{
  __shared__ float srot[8 * 16 * NCOL];

  const int tid  = threadIdx.x;
  const int wave = tid >> 5;
  const int lane = tid & 31;
  const int l16  = lane & 15;
  const int hi   = (lane >> 4) & 1;

  const int bh = blockIdx.y;
  const int b  = bh >> 4;
  const int h  = bh & 15;
  const int mtile = blockIdx.x * 8 + wave;
  const int t     = mtile * 16 + l16;

  const float* aptr = hidden + ((size_t)(b * TT + t)) * HD + h * DD + 2 * hi;
  const float* bptr = rot + (size_t)h * DD * NCOL + (2 * hi) * NCOL;

  v8f acc[8];
#pragma unroll
  for (int nt = 0; nt < 8; ++nt) acc[nt] = {};

#pragma unroll
  for (int s = 0; s < 16; ++s) {
    const v2f a = *(const v2f*)(aptr + 4 * s);
    const float* bp = bptr + (size_t)(4 * s) * NCOL;
#pragma unroll
    for (int nt = 0; nt < 8; ++nt) {
      v2f bv;
      bv[0] = bp[nt * 16 + l16];
      bv[1] = bp[nt * 16 + l16 + NCOL];
      acc[nt] = __builtin_amdgcn_wmma_f32_16x16x4_f32(
          false, a, false, bv, (short)0, acc[nt], false, false);
    }
  }

  float* sl = srot + wave * (16 * NCOL);
#pragma unroll
  for (int nt = 0; nt < 8; ++nt)
#pragma unroll
    for (int v = 0; v < 8; ++v)
      sl[(v + 8 * hi) * NCOL + nt * 16 + l16] = acc[nt][v];

  __syncthreads();

  const float* rowp = sl + l16 * NCOL;
  unsigned int bpair = 0;
#pragma unroll
  for (int nn = 0; nn < 2; ++nn) {
    const int n = 2 * hi + nn;
    const float* xp = rowp + n * 32;
    float bestv = xp[0];
    int   besti = 0;
#pragma unroll 4
    for (int r = 1; r < 32; ++r) {
      float v = xp[r];
      if (v > bestv) { bestv = v; besti = r; }
    }
#pragma unroll 4
    for (int r = 0; r < 32; ++r) {
      float v = -xp[r];
      if (v > bestv) { bestv = v; besti = 32 + r; }
    }
    bpair |= (unsigned int)besti << (8 * n);
  }
  const unsigned int other  = __shfl_xor(bpair, 16, 32);
  const unsigned int packed = bpair | other;
  if (!hi) packed_buckets[bh * TT + mtile * 16 + l16] = packed;
}

__global__ __launch_bounds__(256) void lsh_mask_kernel(
    const unsigned int* __restrict__ packed,
    float* __restrict__ out)
{
  __shared__ unsigned int sb[TT];

  const int tid = threadIdx.x;
  const int bh  = blockIdx.y;
  for (int i = tid; i < TT; i += 256) sb[i] = packed[bh * TT + i];
  __syncthreads();

  const int wave = tid >> 5;
  const int lane = tid & 31;
  const int l16  = lane & 15;
  const unsigned int hi = (lane >> 4) & 1u;

  const int slot     = blockIdx.x * 8 + wave;
  const int rowgroup = slot >> 1;
  const int colhalf  = slot & 1;

  v8i A[4][4];
#pragma unroll
  for (int mt = 0; mt < 4; ++mt) {
    const unsigned int rp = sb[rowgroup * 64 + mt * 16 + l16];
#pragma unroll
    for (int n = 0; n < 4; ++n) {
      const unsigned int bb    = (rp >> (8 * n)) & 63u;
      const unsigned int match = (((bb >> 3) & 1u) == hi);
      const unsigned int v     = ((bb >> 5) << 2) | (((bb >> 4) & 1u) << 1) |
                                 ((bb >> 2) & 1u);
      const unsigned int val   = match ? (1u << ((bb & 3u) * 8u)) : 0u;
      v8i a;
#pragma unroll
      for (int k = 0; k < 8; ++k) a[k] = (v == (unsigned int)k) ? (int)val : 0;
      A[mt][n] = a;
    }
  }

  const size_t obase = (size_t)bh * TT * TT;
  const int ct0 = colhalf * 64;
  for (int ct = ct0; ct < ct0 + 64; ++ct) {
    const unsigned int cp = sb[ct * 16 + l16];
    v8i Bm[4];
#pragma unroll
    for (int n = 0; n < 4; ++n) {
      const unsigned int bb    = (cp >> (8 * n)) & 63u;
      const unsigned int match = (((bb >> 3) & 1u) == hi);
      const unsigned int v     = ((bb >> 5) << 2) | (((bb >> 4) & 1u) << 1) |
                                 ((bb >> 2) & 1u);
      const unsigned int val   = match ? (1u << ((bb & 3u) * 8u)) : 0u;
      v8i bvec;
#pragma unroll
      for (int k = 0; k < 8; ++k)
        bvec[k] = (v == (unsigned int)k) ? (int)val : 0;
      Bm[n] = bvec;
    }
#pragma unroll
    for (int mt = 0; mt < 4; ++mt) {
      v8i acc = {};
#pragma unroll
      for (int n = 0; n < 4; ++n)
        acc = __builtin_amdgcn_wmma_i32_16x16x64_iu8(
            false, A[mt][n], false, Bm[n], acc, false, false);
      const int ibase = rowgroup * 64 + mt * 16 + (int)hi * 8;
#pragma unroll
      for (int v = 0; v < 8; ++v)
        out[obase + (size_t)(ibase + v) * TT + ct * 16 + l16] =
            (acc[v] > 0) ? 1.0f : 0.0f;
    }
  }
}

extern "C" void kernel_launch(void* const* d_in, const int* in_sizes, int n_in,
                              void* d_out, int out_size, void* d_ws, size_t ws_size,
                              hipStream_t stream) {
  const float* hidden = (const float*)d_in[0];
  const float* rot    = (const float*)d_in[1];
  unsigned int* packed = (unsigned int*)d_ws;
  float* out = (float*)d_out;

  dim3 b(256);
  dim3 g1(16, 32);
  lsh_rotate_bucket_kernel<<<g1, b, 0, stream>>>(hidden, rot, packed);

  dim3 g2(8, 32);
  lsh_mask_kernel<<<g2, b, 0, stream>>>(packed, out);
}
